// SelfAttention_56779467653400
// MI455X (gfx1250) — matrix operand order repaired and checked by reading, not compiled, not run on hardware
//
#include <hip/hip_runtime.h>
#include <hip/hip_bf16.h>

// ---------------------------------------------------------------------------
// Problem constants (from reference): B=4, H=W=64 -> N=4096, C=512, DK=64
// ---------------------------------------------------------------------------
#define BATCH 4
#define NPOS  4096
#define CDIM  512
#define DKDIM 64
#define MTOT  (BATCH * NPOS)   // 16384 rows total across batches

typedef _Float16 v8h  __attribute__((ext_vector_type(8)));
typedef _Float16 v16h __attribute__((ext_vector_type(16)));
typedef float    v8f  __attribute__((ext_vector_type(8)));

static __device__ inline v8f zero8() {
    v8f z = {0.f, 0.f, 0.f, 0.f, 0.f, 0.f, 0.f, 0.f};
    return z;
}

static __device__ inline v8f wmma_f16(v16h a, v16h b, v8f c) {
    // D = A(16x32 f16) * B(32x16 f16) + C(16x16 f32)
    return __builtin_amdgcn_wmma_f32_16x16x32_f16(
        /*neg_a=*/false, a, /*neg_b=*/false, b,
        /*c_mod=*/(short)0, c, /*reuse_a=*/false, /*reuse_b=*/false);
}

// A-fragment loader: A is 16x32 (MxK) f16, row-major with leading dim `ld`
// (halves). Layout used here: lane m = lane&15; kbase = (lane>>4)*16;
// element e -> K = k0+kbase+e (contiguous, matches B fragment).
static __device__ inline v16h load_a_frag(const _Float16* base,
                                          int lane, int k0, int ld) {
    const int m  = lane & 15;
    const int kb = (lane >> 4) * 16;
    const _Float16* p = base + (size_t)m * ld + k0 + kb;
    v8h lo = *reinterpret_cast<const v8h*>(p);        // 16B aligned
    v8h hi = *reinterpret_cast<const v8h*>(p + 8);    // 16B aligned
    v16h r;
#pragma unroll
    for (int i = 0; i < 8; ++i) { r[i] = lo[i]; r[i + 8] = hi[i]; }
    return r;
}

// B-fragment loader: B is 32x16 (KxN) f16 stored "n-major": value B[k][n] at
// base[n*ld + k]. Layout used here: lane n = lane&15; kbase = (lane>>4)*16;
// element e -> K = k0+kbase+e  (16 contiguous halves -> one 32B load).
static __device__ inline v16h load_b_frag(const _Float16* base,
                                          int lane, int k0, int ld) {
    const int n  = lane & 15;
    const int kb = (lane >> 4) * 16;
    return *reinterpret_cast<const v16h*>(base + (size_t)n * ld + k0 + kb);
}

// ---------------------------------------------------------------------------
// CDNA5 async global->LDS copy (ASYNCcnt path) + wait. Inline asm per the
// bridge doc (portable across both toolchains). LDS byte address = low 32
// bits of the generic pointer (ISA 10.2 LDS aperture rule).
// ---------------------------------------------------------------------------
static __device__ inline void async_copy_b128(const _Float16* g, _Float16* l) {
    unsigned lds_off = (unsigned)(size_t)l;
    asm volatile("global_load_async_to_lds_b128 %0, %1, off"
                 :: "v"(lds_off), "v"(g) : "memory");
}
static __device__ inline void wait_async0() {
    asm volatile("s_wait_asynccnt 0x0" ::: "memory");
}

// ---------------------------------------------------------------------------
// fp32 -> fp16 conversion / weight transposition (tiny, bandwidth-trivial)
// ---------------------------------------------------------------------------
__global__ void cvt_f32_to_f16(const float* __restrict__ in,
                               _Float16* __restrict__ out, int n) {
    int i = blockIdx.x * blockDim.x + threadIdx.x;
    int stride = gridDim.x * blockDim.x;
    for (; i < n; i += stride) out[i] = (_Float16)in[i];
}

// w: [K][Ncols] f32  ->  wt: [Ncols][K] f16
__global__ void transpose_w_f16(const float* __restrict__ w,
                                _Float16* __restrict__ wt, int K, int Ncols) {
    int i = blockIdx.x * blockDim.x + threadIdx.x;
    int total = K * Ncols;
    int stride = gridDim.x * blockDim.x;
    for (; i < total; i += stride) {
        int k = i / Ncols;
        int n = i - k * Ncols;
        wt[(size_t)n * K + k] = (_Float16)w[i];
    }
}

// ---------------------------------------------------------------------------
// Projection GEMM:  out[m,n] = (sum_k A[m,k]*W[k,n] + bias[n]) * scale
//   A  : [MTOT][512] f16 ; Bt : [Nout][512] f16 (transposed weights)
// Block: 8 waves, 128 rows x 64 cols. The 64x512 weight panel is staged
// once per block into LDS with async b128 copies; each wave register-blocks
// 16 rows x 64 cols (A-frag reused by 4 WMMAs per k-step).
//   store_transposed==0 : out row-major [MTOT][Nout]           (Q, K)
//   store_transposed==1 : out = vt, layout [B][512][4096]       (V)
// ---------------------------------------------------------------------------
#define WPAD 520   // padded K-stride (halves) of LDS weight panel (1040B rows)

__global__ __launch_bounds__(256) void gemm_proj(
    const _Float16* __restrict__ A,
    const _Float16* __restrict__ Bt,
    const float* __restrict__ bias,
    _Float16* __restrict__ out,
    int Nout, float scale, int store_transposed) {

    __shared__ __align__(16) _Float16 Wp[64 * WPAD];   // ~66.6 KB

    const int lane = threadIdx.x & 31;
    const int wave = threadIdx.x >> 5;
    const int row0 = (blockIdx.x * 8 + wave) * 16;
    const int n0   = blockIdx.y * 64;

    // ---- async-stage the 64-col x 512-k f16 weight panel ----------------
    {
        const int col = threadIdx.x >> 2;     // 0..63
        const int q4  = threadIdx.x & 3;      // 256B quarter of a column
        const _Float16* g = Bt + (size_t)(n0 + col) * CDIM + q4 * 128;
        _Float16* l = &Wp[col * WPAD + q4 * 128];
#pragma unroll
        for (int i = 0; i < 16; ++i)
            async_copy_b128(g + i * 8, l + i * 8);
    }
    wait_async0();
    __syncthreads();

    const _Float16* Abase = A + (size_t)row0 * CDIM;
    v8f acc[4];
#pragma unroll
    for (int j = 0; j < 4; ++j) acc[j] = zero8();

#pragma unroll 2
    for (int k0 = 0; k0 < CDIM; k0 += 32) {
        v16h a = load_a_frag(Abase, lane, k0, CDIM);
#pragma unroll
        for (int j = 0; j < 4; ++j) {
            v16h b = load_b_frag(&Wp[(j * 16) * WPAD], lane, k0, WPAD);
            acc[j] = wmma_f16(a, b, acc[j]);
        }
    }

    const int n    = lane & 15;
    const int half = lane >> 4;
#pragma unroll
    for (int j = 0; j < 4; ++j) {
        const int col = n0 + j * 16 + n;
        const float bv = bias[col];
        if (!store_transposed) {
            _Float16* o = out + (size_t)(row0 + half * 8) * Nout + col;
#pragma unroll
            for (int r = 0; r < 8; ++r)
                o[(size_t)r * Nout] = (_Float16)((acc[j][r] + bv) * scale);
        } else {
            const int bidx = row0 >> 12;                 // row0 / 4096
            const int rloc = (row0 & (NPOS - 1)) + half * 8;
            v8h vals;
#pragma unroll
            for (int r = 0; r < 8; ++r)
                vals[r] = (_Float16)((acc[j][r] + bv) * scale);
            *reinterpret_cast<v8h*>(
                out + ((size_t)bidx * CDIM + col) * NPOS + rloc) = vals;
        }
    }
}

// ---------------------------------------------------------------------------
// Flash attention, QTILE = 32 rows / workgroup, key chunks of 64.
//   S phase : wave w -> rowgroup rg=w&1 (16 rows), key subtile cs=w>>1.
//   PV phase: wave w -> output cols [w*64, w*64+64), BOTH rowgroups
//             (each V B-fragment feeds 2 WMMAs -> halves V fragment traffic).
//   Softmax : every wave keeps fp32 running max/sum for all 32 rows,
//             combined from per-wave partials exchanged via LDS.
// ---------------------------------------------------------------------------
#define QTILE 32
#define KT    64
#define PSTR  72    // P_lds row stride in halves (144B, 16B-aligned)

__global__ __launch_bounds__(256) void attn_flash(
    const _Float16* __restrict__ qh,   // [B][4096][64], pre-scaled by 1/8
    const _Float16* __restrict__ kh,   // [B][4096][64]
    const _Float16* __restrict__ vt,   // [B][512][4096] (transposed V)
    const float* __restrict__ x,       // [B][4096][512]
    const float* __restrict__ gamma,   // [1]
    float* __restrict__ out) {         // [B][4096][512]

    __shared__ __align__(16) _Float16 Pl[QTILE * PSTR];
    __shared__ float mbuf[8][16];
    __shared__ float sbuf[8][16];

    const int lane = threadIdx.x & 31;
    const int wave = threadIdx.x >> 5;
    const int half = lane >> 4;
    const int l15  = lane & 15;
    const int rg   = wave & 1;     // S-phase row group
    const int cs   = wave >> 1;    // S-phase key subtile
    const int b    = blockIdx.x >> 7;              // / 128
    const int row0 = (blockIdx.x & 127) * QTILE;

    const _Float16* kbp = kh + (size_t)b * NPOS * DKDIM;
    const _Float16* vb  = vt + (size_t)b * CDIM * NPOS;

    // Q fragments for this wave's S row group (resident all kernel)
    const _Float16* qb = qh + ((size_t)b * NPOS + row0 + rg * 16) * DKDIM;
    const v16h aq0 = load_a_frag(qb, lane, 0,  DKDIM);
    const v16h aq1 = load_a_frag(qb, lane, 32, DKDIM);

    float mrow[2][8], lsum[2][8];
    v8f acc[2][4];
#pragma unroll
    for (int g = 0; g < 2; ++g)
#pragma unroll
        for (int r = 0; r < 8; ++r) { mrow[g][r] = -3.0e38f; lsum[g][r] = 0.f; }
#pragma unroll
    for (int g = 0; g < 2; ++g)
#pragma unroll
        for (int j = 0; j < 4; ++j) acc[g][j] = zero8();

    for (int key0 = 0; key0 < NPOS; key0 += KT) {
        // ---- S tile: rows of rowgroup rg, keys key0 + cs*16 ------------
        const _Float16* kt = kbp + (size_t)(key0 + cs * 16) * DKDIM;
        v8f s = zero8();
        s = wmma_f16(aq0, load_b_frag(kt, lane, 0,  DKDIM), s);
        s = wmma_f16(aq1, load_b_frag(kt, lane, 32, DKDIM), s);

        // ---- per-row max of this 16x16 tile ----------------------------
        float rm[8];
#pragma unroll
        for (int r = 0; r < 8; ++r) {
            float v = s[r];
#pragma unroll
            for (int m = 1; m <= 8; m <<= 1) v = fmaxf(v, __shfl_xor(v, m, 32));
            rm[r] = v;
        }
        if (l15 == 0) {
#pragma unroll
            for (int r = 0; r < 8; ++r) mbuf[wave][half * 8 + r] = rm[r];
        }
        __syncthreads();

        // ---- new maxima + rescale factors for BOTH rowgroups -----------
        float scl[2][8], nmy[8];
#pragma unroll
        for (int g = 0; g < 2; ++g) {
#pragma unroll
            for (int r = 0; r < 8; ++r) {
                float v = mrow[g][r];
#pragma unroll
                for (int c2 = 0; c2 < 4; ++c2)
                    v = fmaxf(v, mbuf[g + 2 * c2][half * 8 + r]);
                scl[g][r] = __expf(mrow[g][r] - v);   // 0 on first chunk
                mrow[g][r] = v;
                if (g == rg) nmy[r] = v;
            }
        }

        // ---- P = exp(S - max), partial row sums, stage P in LDS --------
        float ps[8];
#pragma unroll
        for (int r = 0; r < 8; ++r) {
            float p = __expf(s[r] - nmy[r]);
            float t = p;
#pragma unroll
            for (int m = 1; m <= 8; m <<= 1) t += __shfl_xor(t, m, 32);
            ps[r] = t;
            Pl[(rg * 16 + half * 8 + r) * PSTR + cs * 16 + l15] = (_Float16)p;
        }
        if (l15 == 0) {
#pragma unroll
            for (int r = 0; r < 8; ++r) sbuf[wave][half * 8 + r] = ps[r];
        }
        __syncthreads();

        // ---- denominators + accumulator rescale (both rowgroups) -------
#pragma unroll
        for (int g = 0; g < 2; ++g) {
#pragma unroll
            for (int r = 0; r < 8; ++r) {
                float t = 0.f;
#pragma unroll
                for (int c2 = 0; c2 < 4; ++c2)
                    t += sbuf[g + 2 * c2][half * 8 + r];
                lsum[g][r] = lsum[g][r] * scl[g][r] + t;
#pragma unroll
                for (int j = 0; j < 4; ++j)
                    acc[g][j][r] = acc[g][j][r] * scl[g][r];
            }
        }

        // ---- acc += P(32x64) * V(64 x 64-cols-of-this-wave) ------------
#pragma unroll
        for (int kk = 0; kk < 2; ++kk) {
            v16h pa[2];
#pragma unroll
            for (int g = 0; g < 2; ++g) {
                const _Float16* pp =
                    &Pl[(g * 16 + l15) * PSTR + kk * 32 + half * 16];
                v8h lo = *reinterpret_cast<const v8h*>(pp);
                v8h hi = *reinterpret_cast<const v8h*>(pp + 8);
#pragma unroll
                for (int i = 0; i < 8; ++i) { pa[g][i] = lo[i]; pa[g][i + 8] = hi[i]; }
            }
#pragma unroll
            for (int j = 0; j < 4; ++j) {
                const int col0 = wave * 64 + j * 16;
                v16h bvf = load_b_frag(vb + (size_t)col0 * NPOS + key0,
                                       lane, kk * 32, NPOS);
                acc[0][j] = wmma_f16(pa[0], bvf, acc[0][j]);   // B reused x2
                acc[1][j] = wmma_f16(pa[1], bvf, acc[1][j]);
            }
        }
        __syncthreads();   // protect Pl/mbuf/sbuf before next chunk
    }

    // ---- epilogue: out = gamma * (acc / lsum) + x ----------------------
    const float gma = gamma[0];
#pragma unroll
    for (int g = 0; g < 2; ++g) {
#pragma unroll
        for (int j = 0; j < 4; ++j) {
            const int col = wave * 64 + j * 16 + l15;
#pragma unroll
            for (int r = 0; r < 8; ++r) {
                const int row = row0 + g * 16 + half * 8 + r;
                const size_t idx = ((size_t)b * NPOS + row) * CDIM + col;
                out[idx] = gma * (acc[g][j][r] / lsum[g][r]) + x[idx];
            }
        }
    }
}

// ---------------------------------------------------------------------------
// Host-side launch
// ---------------------------------------------------------------------------
extern "C" void kernel_launch(void* const* d_in, const int* in_sizes, int n_in,
                              void* d_out, int out_size, void* d_ws, size_t ws_size,
                              hipStream_t stream) {
    const float* x     = (const float*)d_in[0];
    const float* wq    = (const float*)d_in[1];
    const float* bq    = (const float*)d_in[2];
    const float* wk    = (const float*)d_in[3];
    const float* bk    = (const float*)d_in[4];
    const float* wv    = (const float*)d_in[5];
    const float* bv    = (const float*)d_in[6];
    const float* gamma = (const float*)d_in[7];
    float* out = (float*)d_out;

    // Workspace carve-up (all 256B aligned)
    char* ws = (char*)d_ws;
    size_t off = 0;
    auto carve = [&](size_t bytes) {
        void* p = ws + off;
        off = (off + bytes + 255) & ~(size_t)255;
        return p;
    };
    _Float16* xh  = (_Float16*)carve((size_t)MTOT * CDIM * 2);   // 16 MB
    _Float16* wqt = (_Float16*)carve((size_t)DKDIM * CDIM * 2);  // 64 KB
    _Float16* wkt = (_Float16*)carve((size_t)DKDIM * CDIM * 2);  // 64 KB
    _Float16* wvt = (_Float16*)carve((size_t)CDIM * CDIM * 2);   // 512 KB
    _Float16* qh  = (_Float16*)carve((size_t)MTOT * DKDIM * 2);  // 2 MB
    _Float16* khb = (_Float16*)carve((size_t)MTOT * DKDIM * 2);  // 2 MB
    _Float16* vtb = (_Float16*)carve((size_t)MTOT * CDIM * 2);   // 16 MB
    (void)ws_size; (void)n_in; (void)in_sizes; (void)out_size;

    // 1) fp32 -> fp16 conversions / weight transposes
    cvt_f32_to_f16<<<2048, 256, 0, stream>>>(x, xh, MTOT * CDIM);
    transpose_w_f16<<<64,  256, 0, stream>>>(wq, wqt, CDIM, DKDIM);
    transpose_w_f16<<<64,  256, 0, stream>>>(wk, wkt, CDIM, DKDIM);
    transpose_w_f16<<<512, 256, 0, stream>>>(wv, wvt, CDIM, CDIM);

    // 2) projections (softmax 1/sqrt(DK)=0.125 folded into Q)
    dim3 gq(MTOT / (16 * 8), DKDIM / 64);   // (128, 1)
    gemm_proj<<<gq, 256, 0, stream>>>(xh, wqt, bq, qh,  DKDIM, 0.125f, 0);
    gemm_proj<<<gq, 256, 0, stream>>>(xh, wkt, bk, khb, DKDIM, 1.0f,   0);
    dim3 gv(MTOT / (16 * 8), CDIM / 64);    // (128, 8)
    gemm_proj<<<gv, 256, 0, stream>>>(xh, wvt, bv, vtb, CDIM,  1.0f,   1);

    // 3) flash attention + residual
    attn_flash<<<BATCH * (NPOS / QTILE), 256, 0, stream>>>(
        qh, khb, vtb, x, gamma, out);
}
